// DLRM_87342454932187
// MI455X (gfx1250) — compile-verified, matrix operand order checked by reading, not run on hardware
//
#include <hip/hip_runtime.h>
#include <hip/hip_bf16.h>
#include <math.h>

// ---------------- types / constants ----------------
typedef __attribute__((ext_vector_type(16))) _Float16 v16h;
typedef __attribute__((ext_vector_type(8)))  float    v8f;

union H16 { v16h v; _Float16 h[16]; };
union F8  { v8f  v; float    f[8];  };

#define DD  28      // embedding dim
#define DP  32      // padded dim (one WMMA K-step)
#define LL  100     // history length
#define LP  112     // padded L (7 x 16 tiles)
#define LK  128     // padded L for K-dim of ctx GEMM
#define BB  4096    // batch
#define NU_ 138493
#define NI_ 27278
#define INV_SQRT_D 0.1889822365046136f   // 1/sqrt(28)

// A-fragment (16x32 f16) per-lane K index, per CDNA5 ISA 7.12.2.
__device__ __forceinline__ int wmma_kidx(int lane, int e) {
  int g  = e >> 1;
  int hb = e & 1;
  int kb = (lane & 16) ? 8 : 0;
  return (g < 4) ? (kb + (g << 1) + hb) : (kb + 16 + ((g - 4) << 1) + hb);
}

__device__ __forceinline__ v8f wmma32(v16h a, v16h b, v8f c) {
  return __builtin_amdgcn_wmma_f32_16x16x32_f16(false, a, false, b, (short)0, c,
                                                false, false);
}

// ---------------------------------------------------------------------------
// User-history tower: gather + QKV + causal attention + DIN, fused per batch b.
// grid = BB blocks of 256 threads (8 waves).
// ---------------------------------------------------------------------------
__global__ void __launch_bounds__(256)
utower_kernel(const int* __restrict__ history, const float* __restrict__ hist_ratings,
              const int* __restrict__ hist_ts, const int* __restrict__ sample_ts,
              const float* __restrict__ hist_emb,
              const float* __restrict__ rating_W, const float* __restrict__ rating_b,
              const float* __restrict__ Wq, const float* __restrict__ Wk,
              const float* __restrict__ Wv,
              const float* __restrict__ din_W1, const float* __restrict__ din_b1,
              const float* __restrict__ din_W2, const float* __restrict__ din_b2,
              const float* __restrict__ ue_buf, const float* __restrict__ ie_buf,
              float* __restrict__ u_int)
{
  __shared__ union {
    _Float16 Ah[LP * LK];                                   // attn weights -> DIN A
    struct { float he[LP * DP]; _Float16 w3[3 * DP * DP]; } p1; // phase-1 inputs
    float h[LP * 64];                                       // DIN hidden
  } u1;
  __shared__ union {
    float    ctx[LP * DP];
    _Float16 qk[2 * LP * DP];                               // Q then K (f16)
  } u2;
  __shared__ _Float16 sV[LK * DP];
  __shared__ _Float16 sRE[LP * DP];
  __shared__ float sMask[LP];
  __shared__ float sS[LP];
  __shared__ float sQt[64];
  __shared__ float sBias[64];

  const int b     = blockIdx.x;
  const int tid   = threadIdx.x;
  const int lane  = tid & 31;
  const int wv    = tid >> 5;
  const int col   = lane & 15;
  const int rbase = (lane & 16) ? 8 : 0;

  // ---- phase 0: qt, weight staging, gather he/re/mask --------------------
  if (tid < 64) {
    float q = 0.f;
    if (tid < DD)            q = ie_buf[b * DD + tid];
    else if (tid < 2 * DD)   q = ue_buf[b * DD + (tid - DD)];
    sQt[tid] = q;
  }
  for (int i = tid; i < 3 * DP * DP; i += 256) {
    int mat = i >> 10, r = i & 1023, k = r >> 5, n = r & 31;
    const float* Wsrc = (mat == 0) ? Wq : (mat == 1) ? Wk : Wv;
    u1.p1.w3[i] = (_Float16)((k < DD && n < DD) ? Wsrc[n * DD + k] : 0.f);
  }
  for (int i = tid; i < (LK - LP) * DP; i += 256)
    sV[LP * DP + i] = (_Float16)0.f;

  const int sts = sample_ts[b];
  for (int l = tid; l < LP; l += 256) {
    int idx = 0, ok = 0; float r = 0.f;
    if (l < LL) {
      idx = history[b * LL + l];
      r   = hist_ratings[b * LL + l];
      int ts = hist_ts[b * LL + l];
      ok = (idx != NI_) && (ts < sts);
    }
    sMask[l] = ok ? 1.f : 0.f;
    for (int d = 0; d < DP; ++d) {
      float hv = 0.f, rv = 0.f;
      if (l < LL && d < DD) {
        hv = hist_emb[(size_t)idx * DD + d];
        rv = r * rating_W[d] + rating_b[d];
      }
      u1.p1.he[l * DP + d] = hv;
      sRE[l * DP + d]      = (_Float16)rv;
    }
  }
  __syncthreads();

  // ---- phase 1: Q,K,V = he @ W^T  (42 WMMA tile jobs) --------------------
  for (int job = wv; job < 42; job += 8) {
    int mat = job / 14, rem = job % 14, mt = rem >> 1, nt = rem & 1;
    H16 a, bf;
#pragma unroll
    for (int e = 0; e < 16; ++e) {
      int k   = wmma_kidx(lane, e);
      a.h[e]  = (_Float16)u1.p1.he[(mt * 16 + col) * DP + k];
      bf.h[e] = u1.p1.w3[mat * 1024 + k * DP + nt * 16 + col];
    }
    F8 c;
#pragma unroll
    for (int i = 0; i < 8; ++i) c.f[i] = 0.f;
    c.v = wmma32(a.v, bf.v, c.v);
    _Float16* dst = (mat == 2) ? sV : (u2.qk + mat * LP * DP);
#pragma unroll
    for (int r = 0; r < 8; ++r)
      dst[(mt * 16 + r + rbase) * DP + nt * 16 + col] = (_Float16)c.f[r];
  }
  __syncthreads();

  // ---- phase 2: scores = Q K^T, masked causal softmax in registers -------
  if (wv < 7) {
    const int mt = wv;
    H16 a;
#pragma unroll
    for (int e = 0; e < 16; ++e)
      a.h[e] = u2.qk[(mt * 16 + col) * DP + wmma_kidx(lane, e)];
    F8 c[7];
    for (int nt = 0; nt < 7; ++nt) {
#pragma unroll
      for (int i = 0; i < 8; ++i) c[nt].f[i] = 0.f;
      H16 bf;
#pragma unroll
      for (int e = 0; e < 16; ++e)
        bf.h[e] = u2.qk[LP * DP + (nt * 16 + col) * DP + wmma_kidx(lane, e)];
      c[nt].v = wmma32(a.v, bf.v, c[nt].v);
    }
#pragma unroll
    for (int r = 0; r < 8; ++r) {
      int m = mt * 16 + r + rbase;
      float v[7]; int ok[7];
      float mx = -3.0e38f;
#pragma unroll
      for (int nt = 0; nt < 7; ++nt) {
        int j  = nt * 16 + col;
        ok[nt] = (sMask[j] > 0.5f) && (j <= m);
        v[nt]  = ok[nt] ? c[nt].f[r] * INV_SQRT_D : -1.0e4f;
        mx     = fmaxf(mx, v[nt]);
      }
      for (int off = 1; off < 16; off <<= 1)   // stays within 16-lane half
        mx = fmaxf(mx, __shfl_xor(mx, off, 32));
      float sum = 0.f;
#pragma unroll
      for (int nt = 0; nt < 7; ++nt) {
        float ev = ok[nt] ? __expf(v[nt] - mx) : 0.f;
        v[nt] = ev; sum += ev;
      }
      for (int off = 1; off < 16; off <<= 1)
        sum += __shfl_xor(sum, off, 32);
      float inv = 1.f / fmaxf(sum, 1e-8f);
#pragma unroll
      for (int nt = 0; nt < 7; ++nt)
        u1.Ah[m * LK + nt * 16 + col] = (_Float16)(v[nt] * inv);
      u1.Ah[m * LK + LP + col] = (_Float16)0.f;   // zero K-pad cols 112..127
    }
  }
  __syncthreads();

  // ---- phase 3: ctx = A @ V  (14 tile jobs, K = 128) ---------------------
  for (int job = wv; job < 14; job += 8) {
    int mt = job >> 1, nt = job & 1;
    F8 c;
#pragma unroll
    for (int i = 0; i < 8; ++i) c.f[i] = 0.f;
    for (int ks = 0; ks < 4; ++ks) {
      H16 a, bf;
#pragma unroll
      for (int e = 0; e < 16; ++e) {
        int k   = wmma_kidx(lane, e);
        a.h[e]  = u1.Ah[(mt * 16 + col) * LK + ks * 32 + k];
        bf.h[e] = sV[(ks * 32 + k) * DP + nt * 16 + col];
      }
      c.v = wmma32(a.v, bf.v, c.v);
    }
#pragma unroll
    for (int r = 0; r < 8; ++r)
      u2.ctx[(mt * 16 + r + rbase) * DP + nt * 16 + col] = c.f[r];
  }
  __syncthreads();

  // ---- phase 4: build DIN A (units | qt*units), effective bias -----------
  for (int l = tid; l < LP; l += 256) {
    for (int cc = 0; cc < 56; ++cc) {
      float u = (cc < DD) ? u2.ctx[l * DP + cc] : (float)sRE[l * DP + (cc - DD)];
      u1.Ah[l * LK + cc]      = (_Float16)u;
      u1.Ah[l * LK + 56 + cc] = (_Float16)(sQt[cc] * u);
    }
    for (int cc = LP; cc < LK; ++cc) u1.Ah[l * LK + cc] = (_Float16)0.f;
  }
  if (tid < 64) {
    float bv = din_b1[tid];
    for (int cc = 0; cc < 56; ++cc) bv += sQt[cc] * din_W1[tid * 168 + cc];
    sBias[tid] = bv;
  }
  __syncthreads();

  // ---- phase 5: h = relu(A @ W1'^T + bias')  (28 tile jobs) --------------
  F8 acc[4];
  int jobids[4], njobs = 0;
  for (int job = wv; job < 28; job += 8) jobids[njobs++] = job;
  for (int i = 0; i < njobs; ++i) {
    int mt = jobids[i] >> 2, nt = jobids[i] & 3;
    F8 c;
#pragma unroll
    for (int q = 0; q < 8; ++q) c.f[q] = 0.f;
    for (int ks = 0; ks < 4; ++ks) {
      H16 a, bf;
#pragma unroll
      for (int e = 0; e < 16; ++e) {
        int k  = wmma_kidx(lane, e);
        int kk = ks * 32 + k;
        a.h[e]  = u1.Ah[(mt * 16 + col) * LK + kk];
        bf.h[e] = (kk < LP) ? (_Float16)din_W1[(nt * 16 + col) * 168 + 56 + kk]
                            : (_Float16)0.f;
      }
      c.v = wmma32(a.v, bf.v, c.v);
    }
    acc[i] = c;
  }
  __syncthreads();
  for (int i = 0; i < njobs; ++i) {
    int mt = jobids[i] >> 2, nt = jobids[i] & 3;
    int n = nt * 16 + col;
#pragma unroll
    for (int r = 0; r < 8; ++r)
      u1.h[(mt * 16 + r + rbase) * 64 + n] = fmaxf(acc[i].f[r] + sBias[n], 0.f);
  }
  __syncthreads();

  // ---- phase 6: s, masked softmax, u_int ---------------------------------
  for (int l = tid; l < LP; l += 256) {
    float s = din_b2[0];
    for (int n = 0; n < 64; ++n) s += u1.h[l * 64 + n] * din_W2[n];
    sS[l] = s;
  }
  __syncthreads();
  if (tid < 32) {
    float mx = -3.0e38f;
    for (int l = lane; l < LP; l += 32)
      mx = fmaxf(mx, (sMask[l] > 0.5f) ? sS[l] : -1.0e4f);
    for (int off = 1; off < 32; off <<= 1) mx = fmaxf(mx, __shfl_xor(mx, off, 32));
    float sum = 0.f;
    for (int l = lane; l < LP; l += 32) {
      float ev = (sMask[l] > 0.5f) ? __expf(sS[l] - mx) : 0.f;
      sS[l] = ev; sum += ev;
    }
    for (int off = 1; off < 32; off <<= 1) sum += __shfl_xor(sum, off, 32);
    float inv = 1.f / fmaxf(sum, 1e-8f);
    for (int l = lane; l < LP; l += 32) sS[l] *= inv;
  }
  __syncthreads();
  if (tid < DD) {
    float a = 0.f;
    for (int l = 0; l < LP; ++l)
      a += sS[l] * (u2.ctx[l * DP + tid] + (float)sRE[l * DP + tid]);
    u_int[b * DD + tid] = a;
  }
}

// ---------------------------------------------------------------------------
// Item-history tower: DIN over past raters (no attention). grid = BB x 256.
// ---------------------------------------------------------------------------
__global__ void __launch_bounds__(256)
itower_kernel(const int* __restrict__ item_hist, const float* __restrict__ ih_ratings,
              const int* __restrict__ ih_ts, const int* __restrict__ sample_ts,
              const float* __restrict__ rater_emb,
              const float* __restrict__ rating_W, const float* __restrict__ rating_b,
              const float* __restrict__ W1, const float* __restrict__ b1,
              const float* __restrict__ W2, const float* __restrict__ b2,
              const float* __restrict__ ue_buf, const float* __restrict__ ie_buf,
              float* __restrict__ i_int)
{
  __shared__ union { _Float16 A[LP * LK]; float h[LP * 64]; } u1;
  __shared__ float sBase[LP * DP];
  __shared__ float sMask[LP];
  __shared__ float sS[LP];
  __shared__ float sQt[64];
  __shared__ float sBias[64];

  const int b     = blockIdx.x;
  const int tid   = threadIdx.x;
  const int lane  = tid & 31;
  const int wv    = tid >> 5;
  const int col   = lane & 15;
  const int rbase = (lane & 16) ? 8 : 0;

  if (tid < 64) {
    float q = 0.f;
    if (tid < DD)            q = ue_buf[b * DD + tid];          // q_i = [ue, ie]
    else if (tid < 2 * DD)   q = ie_buf[b * DD + (tid - DD)];
    sQt[tid] = q;
  }
  __syncthreads();

  const int sts = sample_ts[b];
  for (int l = tid; l < LP; l += 256) {
    int idx = 0, ok = 0; float r = 0.f;
    if (l < LL) {
      idx = item_hist[b * LL + l];
      r   = ih_ratings[b * LL + l];
      int ts = ih_ts[b * LL + l];
      ok = (idx != NU_) && (ts < sts);
    }
    sMask[l] = ok ? 1.f : 0.f;
    for (int d = 0; d < DD; ++d) {
      float pe = 0.f, rv = 0.f;
      if (l < LL) {
        pe = rater_emb[(size_t)idx * DD + d];
        rv = r * rating_W[d] + rating_b[d];
      }
      sBase[l * DP + d]       = pe + rv;
      u1.A[l * LK + d]        = (_Float16)pe;
      u1.A[l * LK + DD + d]   = (_Float16)rv;
      u1.A[l * LK + 56 + d]   = (_Float16)(sQt[d] * pe);
      u1.A[l * LK + 84 + d]   = (_Float16)(sQt[DD + d] * rv);
    }
    for (int d = DD; d < DP; ++d) sBase[l * DP + d] = 0.f;
    for (int cc = LP; cc < LK; ++cc) u1.A[l * LK + cc] = (_Float16)0.f;
  }
  if (tid < 64) {
    float bv = b1[tid];
    for (int cc = 0; cc < 56; ++cc) bv += sQt[cc] * W1[tid * 168 + cc];
    sBias[tid] = bv;
  }
  __syncthreads();

  F8 acc[4];
  int jobids[4], njobs = 0;
  for (int job = wv; job < 28; job += 8) jobids[njobs++] = job;
  for (int i = 0; i < njobs; ++i) {
    int mt = jobids[i] >> 2, nt = jobids[i] & 3;
    F8 c;
#pragma unroll
    for (int q = 0; q < 8; ++q) c.f[q] = 0.f;
    for (int ks = 0; ks < 4; ++ks) {
      H16 a, bf;
#pragma unroll
      for (int e = 0; e < 16; ++e) {
        int k  = wmma_kidx(lane, e);
        int kk = ks * 32 + k;
        a.h[e]  = u1.A[(mt * 16 + col) * LK + kk];
        bf.h[e] = (kk < LP) ? (_Float16)W1[(nt * 16 + col) * 168 + 56 + kk]
                            : (_Float16)0.f;
      }
      c.v = wmma32(a.v, bf.v, c.v);
    }
    acc[i] = c;
  }
  __syncthreads();
  for (int i = 0; i < njobs; ++i) {
    int mt = jobids[i] >> 2, nt = jobids[i] & 3;
    int n = nt * 16 + col;
#pragma unroll
    for (int r = 0; r < 8; ++r)
      u1.h[(mt * 16 + r + rbase) * 64 + n] = fmaxf(acc[i].f[r] + sBias[n], 0.f);
  }
  __syncthreads();

  for (int l = tid; l < LP; l += 256) {
    float s = b2[0];
    for (int n = 0; n < 64; ++n) s += u1.h[l * 64 + n] * W2[n];
    sS[l] = s;
  }
  __syncthreads();
  if (tid < 32) {
    float mx = -3.0e38f;
    for (int l = lane; l < LP; l += 32)
      mx = fmaxf(mx, (sMask[l] > 0.5f) ? sS[l] : -1.0e4f);
    for (int off = 1; off < 32; off <<= 1) mx = fmaxf(mx, __shfl_xor(mx, off, 32));
    float sum = 0.f;
    for (int l = lane; l < LP; l += 32) {
      float ev = (sMask[l] > 0.5f) ? __expf(sS[l] - mx) : 0.f;
      sS[l] = ev; sum += ev;
    }
    for (int off = 1; off < 32; off <<= 1) sum += __shfl_xor(sum, off, 32);
    float inv = 1.f / fmaxf(sum, 1e-8f);
    for (int l = lane; l < LP; l += 32) sS[l] *= inv;
  }
  __syncthreads();
  if (tid < DD) {
    float a = 0.f;
    for (int l = 0; l < LP; ++l) a += sS[l] * sBase[l * DP + tid];
    i_int[b * DD + tid] = a;
  }
}

// ---------------------------------------------------------------------------
// Generic WMMA GEMM:  C[M,N] = act(A[M,K] @ W[N,K]^T + bias)
// grid = (M/16, ceil(N/64)); block = 128 (4 waves, one 16-col tile each).
// act: 0 = none, 1 = relu, 2 = sigmoid.
// ---------------------------------------------------------------------------
__global__ void __launch_bounds__(128)
gemm_kernel(const float* __restrict__ A, int lda,
            const float* __restrict__ W, const float* __restrict__ bias,
            float* __restrict__ C, int ldc, int M, int N, int K, int act)
{
  const int lane  = threadIdx.x & 31;
  const int wv    = threadIdx.x >> 5;
  const int col   = lane & 15;
  const int rbase = (lane & 16) ? 8 : 0;
  const int mt    = blockIdx.x;
  const int n0    = (blockIdx.y * 4 + wv) * 16;
  const int row_a = mt * 16 + col;
  const int n     = n0 + col;

  F8 c;
#pragma unroll
  for (int i = 0; i < 8; ++i) c.f[i] = 0.f;

  const int ksteps = (K + 31) / 32;
  for (int ks = 0; ks < ksteps; ++ks) {
    H16 a, bf;
#pragma unroll
    for (int e = 0; e < 16; ++e) {
      int k = ks * 32 + wmma_kidx(lane, e);
      a.h[e]  = (_Float16)((k < K && row_a < M) ? A[(size_t)row_a * lda + k] : 0.f);
      bf.h[e] = (_Float16)((k < K && n < N)     ? W[(size_t)n * K + k]       : 0.f);
    }
    c.v = wmma32(a.v, bf.v, c.v);
  }

  if (n < N) {
    float bv = bias ? bias[n] : 0.f;
#pragma unroll
    for (int r = 0; r < 8; ++r) {
      int row = mt * 16 + r + rbase;
      if (row < M) {
        float v = c.f[r] + bv;
        if (act == 1)      v = fmaxf(v, 0.f);
        else if (act == 2) v = 1.f / (1.f + __expf(-v));
        C[(size_t)row * ldc + n] = v;
      }
    }
  }
}

// ---------------------------------------------------------------------------
// Small helper kernels
// ---------------------------------------------------------------------------
__global__ void gather_ui_kernel(const int* __restrict__ uid, const int* __restrict__ mid,
                                 const float* __restrict__ uemb, const float* __restrict__ iemb,
                                 float* __restrict__ ue, float* __restrict__ ie)
{
  int b = blockIdx.x, d = threadIdx.x;
  if (d < DD) {
    ue[b * DD + d] = uemb[(size_t)uid[b] * DD + d];
    ie[b * DD + d] = iemb[(size_t)mid[b] * DD + d];
  }
}

__global__ void gfield_kernel(const float* __restrict__ ue, const float* __restrict__ g,
                              const float* __restrict__ gate,
                              const float* __restrict__ ug_W, const float* __restrict__ ug_b,
                              float* __restrict__ gf)
{
  int b = blockIdx.x, d = threadIdx.x;
  float p = (d < DD) ? ue[b * DD + d] * g[b * DD + d] : 0.f;
  for (int off = 1; off < 32; off <<= 1) p += __shfl_xor(p, off, 32);
  float ug = p * INV_SQRT_D;
  if (d < DD)
    gf[b * DD + d] = gate[b * DD + d] * g[b * DD + d] + ug * ug_W[d] + ug_b[d];
}

__global__ void se_kernel(const float* __restrict__ ue, const float* __restrict__ ie,
                          const float* __restrict__ ui, const float* __restrict__ ii,
                          const float* __restrict__ de, const float* __restrict__ ge,
                          const float* __restrict__ gf,
                          const float* __restrict__ W1, const float* __restrict__ b1,
                          const float* __restrict__ W2, const float* __restrict__ b2,
                          float* __restrict__ x)
{
  __shared__ float fld[7 * DD];
  __shared__ float m[7], z[DD], s[7];
  int b = blockIdx.x, t = threadIdx.x;
  const float* src[7] = {ue, ie, ui, ii, de, ge, gf};
  for (int i = t; i < 7 * DD; i += 32) {
    int f = i / DD, d = i % DD;
    fld[i] = src[f][b * DD + d];
  }
  __syncthreads();
  if (t < 7) {
    float a = 0.f;
    for (int d = 0; d < DD; ++d) a += fld[t * DD + d];
    m[t] = a * (1.f / DD);
  }
  __syncthreads();
  if (t < DD) {
    float a = b1[t];
    for (int f = 0; f < 7; ++f) a += m[f] * W1[t * 7 + f];
    z[t] = fmaxf(a, 0.f);
  }
  __syncthreads();
  if (t < 7) {
    float a = b2[t];
    for (int nn = 0; nn < DD; ++nn) a += z[nn] * W2[t * DD + nn];
    s[t] = 1.f / (1.f + __expf(-a));
  }
  __syncthreads();
  for (int i = t; i < 7 * DD; i += 32) {
    int f = i / DD;
    x[(size_t)b * (7 * DD) + i] = fld[i] * s[f];
  }
}

// ---------------------------------------------------------------------------
// Host launch
// ---------------------------------------------------------------------------
extern "C" void kernel_launch(void* const* d_in, const int* in_sizes, int n_in,
                              void* d_out, int out_size, void* d_ws, size_t ws_size,
                              hipStream_t stream)
{
  (void)in_sizes; (void)n_in; (void)out_size; (void)ws_size;

  const int*   user_id    = (const int*)d_in[0];
  const int*   movie_id   = (const int*)d_in[1];
  const float* dense      = (const float*)d_in[2];
  const int*   history    = (const int*)d_in[3];
  const float* hist_rat   = (const float*)d_in[4];
  const int*   hist_ts    = (const int*)d_in[5];
  const int*   sample_ts  = (const int*)d_in[6];
  const float* genres     = (const float*)d_in[7];
  const int*   item_hist  = (const int*)d_in[8];
  const float* ih_rat     = (const float*)d_in[9];
  const int*   ih_ts      = (const int*)d_in[10];
  const float* genome     = (const float*)d_in[11];

  const float* P_user_emb  = (const float*)d_in[12];
  const float* P_item_emb  = (const float*)d_in[13];
  const float* P_hist_emb  = (const float*)d_in[14];
  const float* P_rater_emb = (const float*)d_in[15];
  const float* P_rating_W  = (const float*)d_in[16];
  const float* P_rating_b  = (const float*)d_in[17];
  const float* P_Wq        = (const float*)d_in[18];
  const float* P_Wk        = (const float*)d_in[19];
  const float* P_Wv        = (const float*)d_in[20];
  const float* P_din_W1    = (const float*)d_in[21];
  const float* P_din_b1    = (const float*)d_in[22];
  const float* P_din_W2    = (const float*)d_in[23];
  const float* P_din_b2    = (const float*)d_in[24];
  const float* P_idin_W1   = (const float*)d_in[25];
  const float* P_idin_b1   = (const float*)d_in[26];
  const float* P_idin_W2   = (const float*)d_in[27];
  const float* P_idin_b2   = (const float*)d_in[28];
  const float* P_bot_W1    = (const float*)d_in[29];
  const float* P_bot_b1    = (const float*)d_in[30];
  const float* P_bot_W2    = (const float*)d_in[31];
  const float* P_bot_b2    = (const float*)d_in[32];
  const float* P_gen_W     = (const float*)d_in[33];
  const float* P_gen_b     = (const float*)d_in[34];
  const float* P_gn_W1     = (const float*)d_in[35];
  const float* P_gn_b1     = (const float*)d_in[36];
  const float* P_gn_W2     = (const float*)d_in[37];
  const float* P_gn_b2     = (const float*)d_in[38];
  const float* P_gn_W3     = (const float*)d_in[39];
  const float* P_gn_b3     = (const float*)d_in[40];
  const float* P_gate_W    = (const float*)d_in[41];
  const float* P_gate_b    = (const float*)d_in[42];
  const float* P_ug_W      = (const float*)d_in[43];
  const float* P_ug_b      = (const float*)d_in[44];
  const float* P_se_W1     = (const float*)d_in[45];
  const float* P_se_b1     = (const float*)d_in[46];
  const float* P_se_W2     = (const float*)d_in[47];
  const float* P_se_b2     = (const float*)d_in[48];
  const float* P_top_W1    = (const float*)d_in[49];
  const float* P_top_b1    = (const float*)d_in[50];
  const float* P_top_W2    = (const float*)d_in[51];
  const float* P_top_b2    = (const float*)d_in[52];
  const float* P_top_W3    = (const float*)d_in[53];
  const float* P_top_b3    = (const float*)d_in[54];
  const float* P_top_W4    = (const float*)d_in[55];
  const float* P_top_b4    = (const float*)d_in[56];

  float* ws = (float*)d_ws;
  size_t o = 0;
  float* ue    = ws + o; o += (size_t)BB * DD;
  float* ie    = ws + o; o += (size_t)BB * DD;
  float* uint_ = ws + o; o += (size_t)BB * DD;
  float* iint_ = ws + o; o += (size_t)BB * DD;
  float* dh    = ws + o; o += (size_t)BB * 128;
  float* de    = ws + o; o += (size_t)BB * DD;
  float* ge    = ws + o; o += (size_t)BB * DD;
  float* g1    = ws + o; o += (size_t)BB * 256;
  float* g2    = ws + o; o += (size_t)BB * 64;
  float* gg    = ws + o; o += (size_t)BB * DD;
  float* gate  = ws + o; o += (size_t)BB * DD;
  float* gf    = ws + o; o += (size_t)BB * DD;
  float* xx    = ws + o; o += (size_t)BB * 196;
  float* t1    = ws + o; o += (size_t)BB * 256;
  float* t2    = ws + o; o += (size_t)BB * 128;
  float* t3    = ws + o; o += (size_t)BB * 64;

  gather_ui_kernel<<<BB, 32, 0, stream>>>(user_id, movie_id, P_user_emb, P_item_emb, ue, ie);

  utower_kernel<<<BB, 256, 0, stream>>>(history, hist_rat, hist_ts, sample_ts,
                                        P_hist_emb, P_rating_W, P_rating_b,
                                        P_Wq, P_Wk, P_Wv,
                                        P_din_W1, P_din_b1, P_din_W2, P_din_b2,
                                        ue, ie, uint_);

  itower_kernel<<<BB, 256, 0, stream>>>(item_hist, ih_rat, ih_ts, sample_ts,
                                        P_rater_emb, P_rating_W, P_rating_b,
                                        P_idin_W1, P_idin_b1, P_idin_W2, P_idin_b2,
                                        ue, ie, iint_);

  // dense tower
  gemm_kernel<<<dim3(BB / 16, 2), 128, 0, stream>>>(dense, 17, P_bot_W1, P_bot_b1, dh, 128, BB, 128, 17, 1);
  gemm_kernel<<<dim3(BB / 16, 1), 128, 0, stream>>>(dh, 128, P_bot_W2, P_bot_b2, de, DD, BB, DD, 128, 1);
  // genre tower
  gemm_kernel<<<dim3(BB / 16, 1), 128, 0, stream>>>(genres, 20, P_gen_W, P_gen_b, ge, DD, BB, DD, 20, 1);
  // genome tower
  gemm_kernel<<<dim3(BB / 16, 4), 128, 0, stream>>>(genome, 1128, P_gn_W1, P_gn_b1, g1, 256, BB, 256, 1128, 1);
  gemm_kernel<<<dim3(BB / 16, 1), 128, 0, stream>>>(g1, 256, P_gn_W2, P_gn_b2, g2, 64, BB, 64, 256, 1);
  gemm_kernel<<<dim3(BB / 16, 1), 128, 0, stream>>>(g2, 64, P_gn_W3, P_gn_b3, gg, DD, BB, DD, 64, 0);
  // gate + gfield
  gemm_kernel<<<dim3(BB / 16, 1), 128, 0, stream>>>(gg, DD, P_gate_W, P_gate_b, gate, DD, BB, DD, DD, 2);
  gfield_kernel<<<BB, 32, 0, stream>>>(ue, gg, gate, P_ug_W, P_ug_b, gf);
  // SE reweighting + field concat
  se_kernel<<<BB, 32, 0, stream>>>(ue, ie, uint_, iint_, de, ge, gf,
                                   P_se_W1, P_se_b1, P_se_W2, P_se_b2, xx);
  // top MLP
  gemm_kernel<<<dim3(BB / 16, 4), 128, 0, stream>>>(xx, 196, P_top_W1, P_top_b1, t1, 256, BB, 256, 196, 1);
  gemm_kernel<<<dim3(BB / 16, 2), 128, 0, stream>>>(t1, 256, P_top_W2, P_top_b2, t2, 128, BB, 128, 256, 1);
  gemm_kernel<<<dim3(BB / 16, 1), 128, 0, stream>>>(t2, 128, P_top_W3, P_top_b3, t3, 64, BB, 64, 128, 1);
  gemm_kernel<<<dim3(BB / 16, 1), 128, 0, stream>>>(t3, 64, P_top_W4, P_top_b4, (float*)d_out, 1, BB, 1, 64, 0);
}
